// RMSAttnGQA_9878424781192
// MI455X (gfx1250) — compile-verified, matrix operand order checked by reading, not run on hardware
//
#include <hip/hip_runtime.h>
#include <hip/hip_bf16.h>

// Problem constants (match reference)
#define B_    32
#define KVH_  8
#define G_    4
#define D_    128
#define P_    4096
#define N_    4096     // QH*D
#define KN_   1024     // KVH*D
#define OUTF_ 6144     // N + 2*KN
#define SPLIT_ 8
#define KP_    (P_ / SPLIT_)   // 512 keys per part
#define CHUNKS_ (KP_ / 32)     // 16 chunks of 32 keys

typedef __attribute__((ext_vector_type(16))) __bf16 v16bf;
typedef __attribute__((ext_vector_type(8)))  __bf16 v8bf;
typedef __attribute__((ext_vector_type(8)))  float  v8f;

__device__ __forceinline__ float4 ld4(const float* p) { return *(const float4*)p; }

// Pack 16 fp32 (two groups of 8 consecutive-K values) into the 16-bit A/B
// per-lane layout: elements 0..7 = K base+0..7, elements 8..15 = K base+16..23.
__device__ __forceinline__ v16bf pack16(float4 a, float4 b, float4 c, float4 d, float s) {
  v16bf r;
  r[0]  = (__bf16)(a.x * s); r[1]  = (__bf16)(a.y * s);
  r[2]  = (__bf16)(a.z * s); r[3]  = (__bf16)(a.w * s);
  r[4]  = (__bf16)(b.x * s); r[5]  = (__bf16)(b.y * s);
  r[6]  = (__bf16)(b.z * s); r[7]  = (__bf16)(b.w * s);
  r[8]  = (__bf16)(c.x * s); r[9]  = (__bf16)(c.y * s);
  r[10] = (__bf16)(c.z * s); r[11] = (__bf16)(c.w * s);
  r[12] = (__bf16)(d.x * s); r[13] = (__bf16)(d.y * s);
  r[14] = (__bf16)(d.z * s); r[15] = (__bf16)(d.w * s);
  return r;
}

__device__ __forceinline__ v16bf zero16() {
  v16bf r;
#pragma unroll
  for (int i = 0; i < 16; ++i) r[i] = (__bf16)0.0f;
  return r;
}

__device__ __forceinline__ v8f wmma_bf16(v16bf a, v16bf b, v8f c) {
  return __builtin_amdgcn_wmma_f32_16x16x32_bf16(false, a, false, b, (short)0, c,
                                                 false, false);
}

// ---------------------------------------------------------------------------
// Kernel 1: RMS scale per batch row: rscale[b] = rsqrt(mean(x^2))
// ---------------------------------------------------------------------------
__global__ void rms_kernel(const float* __restrict__ X, float* __restrict__ rscale) {
  const int b = blockIdx.x;
  float s = 0.0f;
  for (int i = threadIdx.x; i < N_; i += blockDim.x) {
    float v = X[(size_t)b * N_ + i];
    s += v * v;
  }
  __shared__ float red[256];
  red[threadIdx.x] = s;
  __syncthreads();
  for (int off = 128; off > 0; off >>= 1) {
    if ((int)threadIdx.x < off) red[threadIdx.x] += red[threadIdx.x + off];
    __syncthreads();
  }
  if (threadIdx.x == 0) rscale[b] = rsqrtf(red[0] / (float)N_);
}

// ---------------------------------------------------------------------------
// Kernel 2: fused QKV projection, out = (x*rscale) @ W^T  (W row-major [out,4096])
// One wave per 16 output features; M=32 (whole batch) via two WMMA C tiles.
// ---------------------------------------------------------------------------
__global__ void qkv_kernel(const float* __restrict__ X,
                           const float* __restrict__ Wq,
                           const float* __restrict__ Wk,
                           const float* __restrict__ Wv,
                           const float* __restrict__ rscale,
                           float* __restrict__ qbuf,
                           float* __restrict__ kbuf,
                           float* __restrict__ vbuf) {
  const int wave = threadIdx.x >> 5;
  const int lane = threadIdx.x & 31;
  const int lrow = lane & 15;
  const int hi   = lane >> 4;
  const int kA   = hi * 8;  // per-lane K base within a 32-wide K step

  const int o0 = (blockIdx.x * 4 + wave) * 16;
  const float* W;
  float* outp;
  int orow, width;
  if (o0 < N_)            { W = Wq; outp = qbuf; orow = o0;         width = N_;  }
  else if (o0 < N_ + KN_) { W = Wk; outp = kbuf; orow = o0 - N_;    width = KN_; }
  else                    { W = Wv; outp = vbuf; orow = o0 - N_ - KN_; width = KN_; }

  const float rs0 = rscale[lrow];
  const float rs1 = rscale[16 + lrow];
  const float* xr0  = X + (size_t)lrow * N_;
  const float* xr1  = X + (size_t)(16 + lrow) * N_;
  const float* wrow = W + (size_t)(orow + lrow) * N_;

  v8f c0 = {};
  v8f c1 = {};
  for (int k0 = 0; k0 < N_; k0 += 32) {
    v16bf a0 = pack16(ld4(xr0 + k0 + kA), ld4(xr0 + k0 + kA + 4),
                      ld4(xr0 + k0 + kA + 16), ld4(xr0 + k0 + kA + 20), rs0);
    v16bf a1 = pack16(ld4(xr1 + k0 + kA), ld4(xr1 + k0 + kA + 4),
                      ld4(xr1 + k0 + kA + 16), ld4(xr1 + k0 + kA + 20), rs1);
    v16bf bt = pack16(ld4(wrow + k0 + kA), ld4(wrow + k0 + kA + 4),
                      ld4(wrow + k0 + kA + 16), ld4(wrow + k0 + kA + 20), 1.0f);
    c0 = wmma_bf16(a0, bt, c0);
    c1 = wmma_bf16(a1, bt, c1);
  }

  const int fcol = orow + lrow;  // C layout: lane -> col N = lane%16
#pragma unroll
  for (int v = 0; v < 8; ++v) {
    outp[(size_t)(v + hi * 8) * width + fcol]      = c0[v];
    outp[(size_t)(16 + v + hi * 8) * width + fcol] = c1[v];
  }
}

// ---------------------------------------------------------------------------
// Kernel 3: attention partials. One wave per (b, kv, part); streams 512 keys
// (part 0 also handles the appended new token as a masked extra chunk).
// V is streamed memory->LDS with gfx1250 async-to-LDS DMA, double-buffered so
// the QK^T WMMAs for chunk ch overlap the DMA for chunk ch+1.
// ---------------------------------------------------------------------------
__global__ void __launch_bounds__(32)
attn_part_kernel(const float* __restrict__ qbuf,
                 const float* __restrict__ kbuf,
                 const float* __restrict__ vbuf,
                 const float* __restrict__ cacheK,
                 const float* __restrict__ cacheV,
                 float* __restrict__ pacc,
                 float* __restrict__ psum) {
  const int part = blockIdx.x % SPLIT_;
  const int kv   = (blockIdx.x / SPLIT_) % KVH_;
  const int b    = blockIdx.x / (SPLIT_ * KVH_);
  const int lane = threadIdx.x;
  const int lrow = lane & 15;
  const int hi   = lane >> 4;
  const int kA   = hi * 8;

  __shared__ __align__(16) float  vldsf[2][32 * 128];  // double-buffered V tile, fp32
  __shared__ __align__(16) __bf16 elds[16 * 32];       // E tile, [query][key], bf16

  // Per-lane LDS byte offsets for the async DMA destination (low 32 bits of a
  // generic pointer to LDS are the LDS byte address).
  const unsigned ldsa0 = (unsigned)(size_t)(&vldsf[0][lane * 128]);
  const unsigned ldsa1 = (unsigned)(size_t)(&vldsf[1][lane * 128]);

  // --- load Q in A layout (M = query-in-group, padded to 16 with zeros) ---
  v16bf qa[4];
  {
    const bool qvalid = (lrow < G_);
    const float* qrow =
        qbuf + (size_t)b * N_ + (size_t)(kv * G_ + (lrow & 3)) * D_;
#pragma unroll
    for (int c = 0; c < 4; ++c) {
      if (qvalid) {
        qa[c] = pack16(ld4(qrow + c * 32 + kA), ld4(qrow + c * 32 + kA + 4),
                       ld4(qrow + c * 32 + kA + 16), ld4(qrow + c * 32 + kA + 20),
                       1.0f);
      } else {
        qa[c] = zero16();
      }
    }
  }

  v16bf ones;
#pragma unroll
  for (int i = 0; i < 16; ++i) ones[i] = (__bf16)1.0f;

  v8f acc[8];
#pragma unroll
  for (int i = 0; i < 8; ++i) acc[i] = (v8f){};
  v8f csum = {};

  // Issue the async V DMA for one 32-key chunk (each lane owns one key row:
  // 128 floats = 32 x b128 transfers across the wave).
  auto issue_v_async = [&](int ch, unsigned ldsa) {
    const float* vr =
        cacheV + (((size_t)b * P_ + (part * KP_ + ch * 32 + lane)) * KVH_ + kv) * D_;
#pragma unroll
    for (int j = 0; j < 128; j += 4) {
      asm volatile("global_load_async_to_lds_b128 %0, %1, off"
                   :
                   : "v"(ldsa + (unsigned)(j * 4)), "v"(vr + j)
                   : "memory");
    }
  };

  // Stage the masked new-token V chunk (key 0 = v_new, keys 1..31 = zero) into
  // an LDS buffer with plain DS stores (fp32).
  auto stage_tail_v = [&](float* dst0) {
    const float* vr = vbuf + (size_t)b * KN_ + (size_t)kv * D_;
    float* dst = dst0 + lane * 128;
#pragma unroll
    for (int j = 0; j < 128; j += 4) {
      float4 x = (lane == 0) ? ld4(vr + j) : make_float4(0, 0, 0, 0);
      *(float4*)(dst + j) = x;
    }
  };

  const int nch = CHUNKS_ + ((part == 0) ? 1 : 0);

  // Prologue: start DMA for chunk 0.
  issue_v_async(0, ldsa0);

  for (int ch = 0; ch < nch; ++ch) {
    const bool tail = (ch == CHUNKS_);
    const int buf = ch & 1;
    const int s0 = part * KP_ + ch * 32;

    // --- kick off V staging for the NEXT chunk into the other buffer ---
    bool next_is_async = false;
    if (ch + 1 < nch) {
      if (ch + 1 == CHUNKS_) {
        stage_tail_v(&vldsf[buf ^ 1][0]);
      } else {
        issue_v_async(ch + 1, (buf ^ 1) ? ldsa1 : ldsa0);
        next_is_async = true;
      }
    }

    // --- scores for two 16-key N-tiles, exp, stage E into LDS ---
#pragma unroll
    for (int t = 0; t < 2; ++t) {
      v8f ce = {};
      if (!tail) {
        const float* kbase =
            cacheK + (((size_t)b * P_ + (s0 + t * 16 + lrow)) * KVH_ + kv) * D_;
        if (ch + 1 < CHUNKS_) {  // prefetch next chunk's K row for this lane
          __builtin_prefetch(kbase + (size_t)32 * KVH_ * D_, 0, 3);
        }
#pragma unroll
        for (int ds = 0; ds < 4; ++ds) {
          v16bf bt = pack16(ld4(kbase + ds * 32 + kA), ld4(kbase + ds * 32 + kA + 4),
                            ld4(kbase + ds * 32 + kA + 16),
                            ld4(kbase + ds * 32 + kA + 20), 1.0f);
          ce = wmma_bf16(qa[ds], bt, ce);
        }
      } else {
        const float* kbase = kbuf + (size_t)b * KN_ + (size_t)kv * D_;
        const bool kvalid = (t == 0) && (lrow == 0);
#pragma unroll
        for (int ds = 0; ds < 4; ++ds) {
          v16bf bt;
          if (kvalid) {
            bt = pack16(ld4(kbase + ds * 32 + kA), ld4(kbase + ds * 32 + kA + 4),
                        ld4(kbase + ds * 32 + kA + 16),
                        ld4(kbase + ds * 32 + kA + 20), 1.0f);
          } else {
            bt = zero16();
          }
          ce = wmma_bf16(qa[ds], bt, ce);
        }
      }
      const bool valid = !tail || ((t == 0) && (lrow == 0));
#pragma unroll
      for (int v = 0; v < 8; ++v) {
        float e = valid ? __expf(ce[v]) : 0.0f;
        elds[(v + hi * 8) * 32 + t * 16 + lrow] = (__bf16)e;
      }
    }

    // --- wait for this chunk's V data (async completes in order: with the
    //     next batch of 32 DMAs in flight, <=32 outstanding means ours landed)
    if (!tail) {
      if (next_is_async) {
        asm volatile("s_wait_asynccnt 0x20" ::: "memory");
      } else {
        asm volatile("s_wait_asynccnt 0x0" ::: "memory");
      }
    }
    asm volatile("s_wait_dscnt 0x0" ::: "memory");  // E tile (and tail V) visible

    // --- read E back in A layout ---
    v16bf ea;
    {
      v8bf lo = *(const v8bf*)&elds[lrow * 32 + kA];
      v8bf h8 = *(const v8bf*)&elds[lrow * 32 + kA + 16];
#pragma unroll
      for (int i = 0; i < 8; ++i) { ea[i] = lo[i]; ea[8 + i] = h8[i]; }
    }

    // --- PV: 8 D-tiles plus a ones-column WMMA that yields row sums of E ---
    csum = wmma_bf16(ea, ones, csum);
#pragma unroll
    for (int dt = 0; dt < 8; ++dt) {
      v16bf vb;
      const float* vcol = &vldsf[buf][dt * 16 + lrow];
#pragma unroll
      for (int j = 0; j < 8; ++j) {
        vb[j]     = (__bf16)vcol[(kA + j) * 128];
        vb[8 + j] = (__bf16)vcol[(kA + 16 + j) * 128];
      }
      acc[dt] = wmma_bf16(ea, vb, acc[dt]);
    }
    // Ensure this buffer's DS reads retired before it is re-targeted by DMA.
    asm volatile("s_wait_dscnt 0x0" ::: "memory");
  }

  // --- write partials: rows g=0..3 live in VGPRs 0..3 of lanes 0..15 ---
  if (hi == 0) {
    float* pa = pacc + (((size_t)(b * KVH_ + kv) * SPLIT_ + part) * G_) * D_;
#pragma unroll
    for (int dt = 0; dt < 8; ++dt)
#pragma unroll
      for (int g = 0; g < G_; ++g)
        pa[(size_t)g * D_ + dt * 16 + lrow] = acc[dt][g];
    if (lrow == 0) {
#pragma unroll
      for (int g = 0; g < G_; ++g)
        psum[((size_t)(b * KVH_ + kv) * SPLIT_ + part) * G_ + g] = csum[g];
    }
  }
}

// ---------------------------------------------------------------------------
// Kernel 4: combine partials and normalize.
// ---------------------------------------------------------------------------
__global__ void attn_reduce_kernel(const float* __restrict__ pacc,
                                   const float* __restrict__ psum,
                                   float* __restrict__ out) {
  const int blk = blockIdx.x;  // b*KVH + kv
  const int kv = blk % KVH_;
  const int b  = blk / KVH_;
  const int t  = threadIdx.x;  // d index 0..127
#pragma unroll
  for (int g = 0; g < G_; ++g) {
    float s = 0.0f, a = 0.0f;
    for (int p = 0; p < SPLIT_; ++p) {
      s += psum[((size_t)blk * SPLIT_ + p) * G_ + g];
      a += pacc[(((size_t)blk * SPLIT_ + p) * G_ + g) * D_ + t];
    }
    out[(size_t)b * N_ + (size_t)(kv * G_ + g) * D_ + t] = a / s;
  }
}

// ---------------------------------------------------------------------------
extern "C" void kernel_launch(void* const* d_in, const int* in_sizes, int n_in,
                              void* d_out, int out_size, void* d_ws, size_t ws_size,
                              hipStream_t stream) {
  (void)in_sizes; (void)n_in; (void)out_size; (void)ws_size;
  const float* X      = (const float*)d_in[0];
  const float* Wq     = (const float*)d_in[1];
  const float* Wk     = (const float*)d_in[2];
  const float* Wv     = (const float*)d_in[3];
  const float* cacheK = (const float*)d_in[4];
  const float* cacheV = (const float*)d_in[5];
  float* out = (float*)d_out;

  float* ws     = (float*)d_ws;
  float* rscale = ws;                         // 32
  float* qbuf   = rscale + 32;                // 32*4096
  float* kbuf   = qbuf + (size_t)B_ * N_;     // 32*1024
  float* vbuf   = kbuf + (size_t)B_ * KN_;    // 32*1024
  float* psum   = vbuf + (size_t)B_ * KN_;    // 32*8*8*4
  float* pacc   = psum + (size_t)B_ * KVH_ * SPLIT_ * G_;  // 32*8*8*4*128

  rms_kernel<<<B_, 256, 0, stream>>>(X, rscale);
  qkv_kernel<<<OUTF_ / (16 * 4), 128, 0, stream>>>(X, Wq, Wk, Wv, rscale,
                                                   qbuf, kbuf, vbuf);
  attn_part_kernel<<<B_ * KVH_ * SPLIT_, 32, 0, stream>>>(qbuf, kbuf, vbuf,
                                                          cacheK, cacheV,
                                                          pacc, psum);
  attn_reduce_kernel<<<B_ * KVH_, D_, 0, stream>>>(pacc, psum, out);
}
